// HyperbolicMLP_25598005084536
// MI455X (gfx1250) — compile-verified, matrix operand order checked by reading, not run on hardware
//
#include <hip/hip_runtime.h>
#include <hip/hip_bf16.h>

typedef unsigned short u16;
typedef __attribute__((ext_vector_type(16))) __bf16        v16bf;
typedef __attribute__((ext_vector_type(8)))  float         v8f;
typedef __attribute__((ext_vector_type(8)))  unsigned int  v8u;
typedef __attribute__((vector_size(16)))     int           vi4;

#define TILE_M 128   // block rows
#define TILE_N 256   // block cols
#define TILE_K 32

#if __has_builtin(__builtin_amdgcn_global_load_async_to_lds_b128)
#define USE_ASYNC 1
#if __has_builtin(__builtin_amdgcn_s_wait_asynccnt)
#define WAIT_ASYNC(n) __builtin_amdgcn_s_wait_asynccnt((short)(n))
#else
#define WAIT_ASYNC(n) asm volatile("s_wait_asynccnt %0" ::"i"(n))
#endif
#else
#define USE_ASYNC 0
#define WAIT_ASYNC(n) do {} while (0)
#endif

// ---------------- helpers ----------------
__device__ __forceinline__ u16 f2bf(float f) {
    unsigned int x = __float_as_uint(f);
    return (u16)((x + 0x7fffu + ((x >> 16) & 1u)) >> 16);  // round-to-nearest-even
}

__device__ __forceinline__ void async_cp16(u16* lds_dst, const u16* gsrc) {
#if USE_ASYNC
    __builtin_amdgcn_global_load_async_to_lds_b128(
        (__attribute__((address_space(1))) vi4*)gsrc,
        (__attribute__((address_space(3))) vi4*)lds_dst, 0, 0);
#else
    *(uint4*)lds_dst = *(const uint4*)gsrc;
#endif
}

__device__ __forceinline__ v16bf load_frag(const u16* p) {
    // 16 bf16 per lane: k 0..7 at p, k 16..23 at p+16 (u16 units) per CDNA5 16-bit A/B layout
    uint4 lo = *(const uint4*)(p);
    uint4 hi = *(const uint4*)(p + 16);
    v8u u;
    u[0] = lo.x; u[1] = lo.y; u[2] = lo.z; u[3] = lo.w;
    u[4] = hi.x; u[5] = hi.y; u[6] = hi.z; u[7] = hi.w;
    return __builtin_bit_cast(v16bf, u);
}

// ---------------- f32 -> bf16 converts ----------------
__global__ __launch_bounds__(256)
void cvt_bf16_vec(const float4* __restrict__ in, uint2* __restrict__ out, int n4) {
    int i = blockIdx.x * 256 + threadIdx.x;
    if (i >= n4) return;
    float4 v = in[i];
    uint2 o;
    o.x = (unsigned)f2bf(v.x) | ((unsigned)f2bf(v.y) << 16);
    o.y = (unsigned)f2bf(v.z) | ((unsigned)f2bf(v.w) << 16);
    out[i] = o;
}

__global__ __launch_bounds__(256)
void cvt_bf16_pad(const float* __restrict__ in, u16* __restrict__ out,
                  int rows_in, int cols, int total_out) {
    int i = blockIdx.x * 256 + threadIdx.x;
    if (i >= total_out) return;
    int r = i / cols;
    int c = i - r * cols;
    float v = (r < rows_in) ? in[r * cols + c] : 0.0f;
    out[i] = f2bf(v);
}

// ---------------- WMMA GEMM: C[M,ldC] = A[M,K] * Bw[N,K]^T ----------------
// 256 threads = 8 waves (2x4); each wave owns a 64x64 sub-tile = 4x4 WMMA frags.
// LDS double-buffered; tiles staged with async global->LDS (ASYNCcnt).
__global__ __launch_bounds__(256)
void gemm_bf16_wmma(const u16* __restrict__ A, const u16* __restrict__ Bw,
                    float* __restrict__ C,
                    int M, int N, int K, int ldC, int Nvalid) {
    __shared__ __align__(16) u16 sA[2 * TILE_M * TILE_K];  // 16 KB
    __shared__ __align__(16) u16 sB[2 * TILE_N * TILE_K];  // 32 KB

    const int tid  = threadIdx.x;
    const int lane = tid & 31;
    const int wave = tid >> 5;
    const int wr   = wave >> 2;        // 0..1 : 64-row group
    const int wc   = wave & 3;         // 0..3 : 64-col group
    const int half = lane >> 4;        // 0/1
    const int l16  = lane & 15;
    const int koff = half * 8;         // k base within frag (u16 units)

    const int blockM = blockIdx.y * TILE_M;
    const int blockN = blockIdx.x * TILE_N;

    // staging: 16B chunks; A = 512 chunks (rows rr, rr+64), B = 1024 chunks (rows rr+{0,64,128,192})
    const int rr = tid >> 2;
    const int kc = (tid & 3) * 8;

    const v8f vzero = {0.f,0.f,0.f,0.f,0.f,0.f,0.f,0.f};
    v8f acc[4][4];
#pragma unroll
    for (int mi = 0; mi < 4; ++mi)
#pragma unroll
        for (int ni = 0; ni < 4; ++ni) acc[mi][ni] = vzero;

    const int nk = K / TILE_K;

#define ISSUE_TILE(KT, BI)                                                                 \
    do {                                                                                   \
        const size_t k0_ = (size_t)(KT) * TILE_K;                                          \
        u16* dA_ = sA + (BI) * (TILE_M * TILE_K);                                          \
        u16* dB_ = sB + (BI) * (TILE_N * TILE_K);                                          \
        async_cp16(dA_ + (rr)      * TILE_K + kc, &A[(size_t)(blockM + rr)       * K + k0_ + kc]); \
        async_cp16(dA_ + (rr + 64) * TILE_K + kc, &A[(size_t)(blockM + rr + 64)  * K + k0_ + kc]); \
        async_cp16(dB_ + (rr)      * TILE_K + kc, &Bw[(size_t)(blockN + rr)      * K + k0_ + kc]); \
        async_cp16(dB_ + (rr + 64) * TILE_K + kc, &Bw[(size_t)(blockN + rr + 64) * K + k0_ + kc]); \
        async_cp16(dB_ + (rr + 128)* TILE_K + kc, &Bw[(size_t)(blockN + rr + 128)* K + k0_ + kc]); \
        async_cp16(dB_ + (rr + 192)* TILE_K + kc, &Bw[(size_t)(blockN + rr + 192)* K + k0_ + kc]); \
    } while (0)

    ISSUE_TILE(0, 0);

    for (int kt = 0; kt < nk; ++kt) {
        const int cur = kt & 1;
        if (kt + 1 < nk) {
            ISSUE_TILE(kt + 1, cur ^ 1);
            WAIT_ASYNC(6);   // first 6 (current tile) complete; async loads retire in order
        } else {
            WAIT_ASYNC(0);
        }
        __syncthreads();     // current tile visible to all waves

        const u16* bufA = sA + cur * (TILE_M * TILE_K);
        const u16* bufB = sB + cur * (TILE_N * TILE_K);

        v16bf af[4], bf[4];
#pragma unroll
        for (int mi = 0; mi < 4; ++mi)
            af[mi] = load_frag(&bufA[(wr * 64 + mi * 16 + l16) * TILE_K + koff]);
#pragma unroll
        for (int ni = 0; ni < 4; ++ni)
            bf[ni] = load_frag(&bufB[(wc * 64 + ni * 16 + l16) * TILE_K + koff]);
#pragma unroll
        for (int mi = 0; mi < 4; ++mi)
#pragma unroll
            for (int ni = 0; ni < 4; ++ni)
                acc[mi][ni] = __builtin_amdgcn_wmma_f32_16x16x32_bf16(
                    false, af[mi], false, bf[ni], (short)0, acc[mi][ni], false, false);

        __syncthreads();     // done reading `cur` before it is refilled next iteration
    }
#undef ISSUE_TILE

    // epilogue: lanes 0-15 -> (M=j, N=lane); lanes 16-31 -> (M=j+8, N=lane-16)
#pragma unroll
    for (int mi = 0; mi < 4; ++mi) {
#pragma unroll
        for (int ni = 0; ni < 4; ++ni) {
            const int rbase = blockM + wr * 64 + mi * 16 + half * 8;
            const int cbase = blockN + wc * 64 + ni * 16 + l16;
            if (cbase < Nvalid) {
#pragma unroll
                for (int j = 0; j < 8; ++j)
                    C[(size_t)(rbase + j) * ldC + cbase] = acc[mi][ni][j];
            }
        }
    }
}

// ---------------- BN column statistics ----------------
__global__ void zero_f32(float* p, int n) {
    int i = blockIdx.x * 256 + threadIdx.x;
    if (i < n) p[i] = 0.0f;
}

__global__ __launch_bounds__(256)
void bn_colstats(const float4* __restrict__ Y4, float* __restrict__ sum,
                 float* __restrict__ sumsq, int Ncols4, int rows_per_chunk) {
    int col4 = blockIdx.x * 256 + threadIdx.x;
    int r0 = blockIdx.y * rows_per_chunk;
    float4 s = {0.f,0.f,0.f,0.f}, q = {0.f,0.f,0.f,0.f};
    for (int r = r0; r < r0 + rows_per_chunk; ++r) {
        float4 v = Y4[(size_t)r * Ncols4 + col4];
        s.x += v.x; s.y += v.y; s.z += v.z; s.w += v.w;
        q.x += v.x*v.x; q.y += v.y*v.y; q.z += v.z*v.z; q.w += v.w*v.w;
    }
    int c = col4 * 4;
    atomicAdd(&sum[c+0], s.x); atomicAdd(&sum[c+1], s.y);
    atomicAdd(&sum[c+2], s.z); atomicAdd(&sum[c+3], s.w);
    atomicAdd(&sumsq[c+0], q.x); atomicAdd(&sumsq[c+1], q.y);
    atomicAdd(&sumsq[c+2], q.z); atomicAdd(&sumsq[c+3], q.w);
}

__global__ void bn_finalize(const float* sum, const float* sumsq,
                            float* mu, float* rstd, int Ncols, float invB) {
    int c = blockIdx.x * 256 + threadIdx.x;
    if (c >= Ncols) return;
    float m = sum[c] * invB;
    float v = sumsq[c] * invB - m * m;
    mu[c] = m;
    rstd[c] = rsqrtf(v + 1e-5f);
}

// ---------------- fused BN+ReLU + hyperbolic row maps ----------------
#define SQRT_C 0.31622776601683794f
#define EPS    1e-8f

// Ncols = 2048; thread t owns contiguous cols [t*8, t*8+8) kept in registers.
__global__ __launch_bounds__(256)
void row_hyp(const float* __restrict__ Y, const float* __restrict__ g,
             const float* __restrict__ be, const float* __restrict__ mu,
             const float* __restrict__ rstd, u16* __restrict__ outb,
             int Ncols, int which) {   // which: 0 = log-map path, 1 = exp-map path
    __shared__ float red[256];
    __shared__ float s_scale;
    const int r = blockIdx.x, t = threadIdx.x;
    const int cb = t * 8;

    float v[8];
    float acc = 0.f;
#pragma unroll
    for (int h = 0; h < 2; ++h) {
        float4 y  = *(const float4*)&Y[(size_t)r * Ncols + cb + h * 4];
        float4 gg = *(const float4*)&g[cb + h * 4];
        float4 bb = *(const float4*)&be[cb + h * 4];
        float4 mm = *(const float4*)&mu[cb + h * 4];
        float4 rs = *(const float4*)&rstd[cb + h * 4];
        float* yp = &y.x; float* gp = &gg.x; float* bp = &bb.x; float* mp = &mm.x; float* rp = &rs.x;
#pragma unroll
        for (int j = 0; j < 4; ++j) {
            float w = gp[j] * (yp[j] - mp[j]) * rp[j] + bp[j];
            w = fmaxf(w, 0.f);
            v[h * 4 + j] = w;
            acc += w * w;
        }
    }
    red[t] = acc;
    __syncthreads();
    for (int o = 128; o > 0; o >>= 1) {
        if (t < o) red[t] += red[t + o];
        __syncthreads();
    }
    if (t == 0) {
        float rn = sqrtf(red[0]);
        float s;
        if (which == 0) {
            // euclidean->poincare then log-map at origin (reference formulas)
            if (rn > 0.f) {
                float scale1 = 0.9f * tanhf(SQRT_C * rn) / (SQRT_C * rn);
                float yn = scale1 * rn;                       // |h_hyp|
                float dn = fmaxf(yn / (1.f + EPS), EPS);
                float lambda = 2.f / (1.f + EPS);
                float s2 = (2.f / (SQRT_C * lambda)) * atanhf(SQRT_C * dn) * (-1.f / (1.f + EPS)) / dn;
                s = scale1 * s2;
            } else s = 0.f;
        } else {
            // exp-map at origin then poincare->euclidean
            float vn = fmaxf(rn, EPS);
            float lambda = 2.f / (1.f + EPS);
            float scale = tanhf(SQRT_C * lambda * vn * 0.5f) / (SQRT_C * vn);
            float s1 = scale / (1.f + EPS);                   // h_hyp = s1 * v
            float n = s1 * rn;
            n = fminf(fmaxf(n, 1e-8f), 1.f - 1e-8f);
            float s2 = atanhf(SQRT_C * n) / (SQRT_C * n);
            s = s1 * s2;
        }
        s_scale = s;
    }
    __syncthreads();
    const float s = s_scale;
    uint4 o;
    o.x = (unsigned)f2bf(s * v[0]) | ((unsigned)f2bf(s * v[1]) << 16);
    o.y = (unsigned)f2bf(s * v[2]) | ((unsigned)f2bf(s * v[3]) << 16);
    o.z = (unsigned)f2bf(s * v[4]) | ((unsigned)f2bf(s * v[5]) << 16);
    o.w = (unsigned)f2bf(s * v[6]) | ((unsigned)f2bf(s * v[7]) << 16);
    *(uint4*)&outb[(size_t)r * Ncols + cb] = o;
}

// ---------------- in-place log_softmax over rows of d_out (adds b3) ----------------
__global__ __launch_bounds__(256)
void logsoftmax_rows(float* __restrict__ out, const float* __restrict__ b3, int N) {
    __shared__ float red[256];
    __shared__ float s_max, s_lse;
    const int r = blockIdx.x, t = threadIdx.x;
    float z[4];
    float m = -3.4e38f;
#pragma unroll
    for (int j = 0; j < 4; ++j) {
        int c = t + j * 256;
        if (c < N) {
            z[j] = out[(size_t)r * N + c] + b3[c];
            m = fmaxf(m, z[j]);
        } else z[j] = -3.4e38f;
    }
    red[t] = m; __syncthreads();
    for (int o = 128; o > 0; o >>= 1) {
        if (t < o) red[t] = fmaxf(red[t], red[t + o]);
        __syncthreads();
    }
    if (t == 0) s_max = red[0];
    __syncthreads();
    const float mx = s_max;
    float s = 0.f;
#pragma unroll
    for (int j = 0; j < 4; ++j) {
        int c = t + j * 256;
        if (c < N) s += expf(z[j] - mx);
    }
    red[t] = s; __syncthreads();
    for (int o = 128; o > 0; o >>= 1) {
        if (t < o) red[t] += red[t + o];
        __syncthreads();
    }
    if (t == 0) s_lse = logf(red[0]);
    __syncthreads();
    const float lg = s_lse;
#pragma unroll
    for (int j = 0; j < 4; ++j) {
        int c = t + j * 256;
        if (c < N) out[(size_t)r * N + c] = z[j] - mx - lg;
    }
}

// ---------------- host orchestration ----------------
extern "C" void kernel_launch(void* const* d_in, const int* in_sizes, int n_in,
                              void* d_out, int out_size, void* d_ws, size_t ws_size,
                              hipStream_t stream) {
    (void)in_sizes; (void)n_in; (void)out_size; (void)ws_size;
    const int B = 8192, D_IN = 1024, H = 2048, D_OUT = 1000, NPAD = 1024;

    const float* x   = (const float*)d_in[0];
    const float* W1  = (const float*)d_in[1];
    // b1 (d_in[2]) absorbed by BN mean subtraction
    const float* g1  = (const float*)d_in[3];
    const float* be1 = (const float*)d_in[4];
    const float* W2  = (const float*)d_in[5];
    // b2 (d_in[6]) absorbed by BN
    const float* g2  = (const float*)d_in[7];
    const float* be2 = (const float*)d_in[8];
    const float* W3  = (const float*)d_in[9];
    const float* b3  = (const float*)d_in[10];
    float* out = (float*)d_out;

    char* ws = (char*)d_ws;
    size_t off = 0;
    u16* xb   = (u16*)(ws + off); off += (size_t)B * D_IN * 2;       // 16 MB
    u16* W1b  = (u16*)(ws + off); off += (size_t)H * D_IN * 2;       //  4 MB
    u16* W2b  = (u16*)(ws + off); off += (size_t)H * H * 2;          //  8 MB
    u16* W3b  = (u16*)(ws + off); off += (size_t)NPAD * H * 2;       //  4 MB (padded)
    float* Y  = (float*)(ws + off); off += (size_t)B * H * 4;        // 64 MB (Y1/Y2 reuse)
    u16* Tb   = (u16*)(ws + off); off += (size_t)B * H * 2;          // 32 MB (T1/T2 reuse)
    float* st_sum = (float*)(ws + off); off += H * 4;
    float* st_sq  = (float*)(ws + off); off += H * 4;
    float* st_mu  = (float*)(ws + off); off += H * 4;
    float* st_rs  = (float*)(ws + off); off += H * 4;

    const dim3 blk(256);
    // 1) convert inputs to bf16
    {
        int n4 = (B * D_IN) / 4;
        cvt_bf16_vec<<<dim3((n4 + 255) / 256), blk, 0, stream>>>((const float4*)x,  (uint2*)xb,  n4);
        n4 = (H * D_IN) / 4;
        cvt_bf16_vec<<<dim3((n4 + 255) / 256), blk, 0, stream>>>((const float4*)W1, (uint2*)W1b, n4);
        n4 = (H * H) / 4;
        cvt_bf16_vec<<<dim3((n4 + 255) / 256), blk, 0, stream>>>((const float4*)W2, (uint2*)W2b, n4);
        int n = NPAD * H;
        cvt_bf16_pad<<<dim3((n + 255) / 256), blk, 0, stream>>>(W3, W3b, D_OUT, H, n);
    }

    // 2) GEMM1: Y = x @ W1^T   (8192x1024 * 1024x2048)
    gemm_bf16_wmma<<<dim3(H / TILE_N, B / TILE_M), blk, 0, stream>>>(
        xb, W1b, Y, B, H, D_IN, H, H);

    // 3) BN1 stats + fused BN+ReLU+euclidean->poincare->log-map  -> Tb (bf16)
    zero_f32<<<dim3((2 * H + 255) / 256), blk, 0, stream>>>(st_sum, 2 * H);
    bn_colstats<<<dim3(H / 1024, 64), blk, 0, stream>>>((const float4*)Y, st_sum, st_sq, H / 4, B / 64);
    bn_finalize<<<dim3(H / 256), blk, 0, stream>>>(st_sum, st_sq, st_mu, st_rs, H, 1.0f / B);
    row_hyp<<<dim3(B), blk, 0, stream>>>(Y, g1, be1, st_mu, st_rs, Tb, H, 0);

    // 4) GEMM2: Y = T1 @ W2^T  (8192x2048 * 2048x2048)
    gemm_bf16_wmma<<<dim3(H / TILE_N, B / TILE_M), blk, 0, stream>>>(
        Tb, W2b, Y, B, H, H, H, H);

    // 5) BN2 stats + fused BN+ReLU+exp-map->euclidean -> Tb (bf16)
    zero_f32<<<dim3((2 * H + 255) / 256), blk, 0, stream>>>(st_sum, 2 * H);
    bn_colstats<<<dim3(H / 1024, 64), blk, 0, stream>>>((const float4*)Y, st_sum, st_sq, H / 4, B / 64);
    bn_finalize<<<dim3(H / 256), blk, 0, stream>>>(st_sum, st_sq, st_mu, st_rs, H, 1.0f / B);
    row_hyp<<<dim3(B), blk, 0, stream>>>(Y, g2, be2, st_mu, st_rs, Tb, H, 1);

    // 6) GEMM3: d_out = T2 @ W3^T (store-masked to N=1000, ldC=1000)
    gemm_bf16_wmma<<<dim3(NPAD / TILE_N, B / TILE_M), blk, 0, stream>>>(
        Tb, W3b, out, B, NPAD, H, D_OUT, D_OUT);

    // 7) log_softmax in place (adds b3)
    logsoftmax_rows<<<dim3(B), blk, 0, stream>>>(out, b3, D_OUT);
}
